// InfinigramSidecarWrapper_6141803233591
// MI455X (gfx1250) — matrix operand order repaired and checked by reading, not compiled, not run on hardware
//
#include <hip/hip_runtime.h>
#include <hip/hip_bf16.h>

// ---------- types ----------
typedef __attribute__((ext_vector_type(16))) __bf16 v16bf;
typedef __attribute__((ext_vector_type(8)))  float  v8f;
typedef unsigned int u32x4 __attribute__((ext_vector_type(4)));
typedef int          i32x4 __attribute__((ext_vector_type(4)));
typedef int          i32x8 __attribute__((ext_vector_type(8)));

#define BB 4
#define SS 512
#define DD 1024
#define VOCAB 32000
#define MASKI 0xFFFFFll

#ifndef __has_builtin
#define __has_builtin(x) 0
#endif
#if defined(__AMDGCN__) && __has_builtin(__builtin_amdgcn_tensor_load_to_lds) && \
    __has_builtin(__builtin_amdgcn_s_wait_tensorcnt)
#define USE_TDM 1
#else
#define USE_TDM 0
#endif

struct Sidecar {
    int valid_len;
    int any_hit;
    int top_tok;
    int slen;
    long long stot;
    float conf;
    int pad_;
    int ptok[8];
    float contrib[8];
};

// ---------- helpers ----------
__device__ __forceinline__ unsigned short f2bf_rne(float f) {
    unsigned u = __float_as_uint(f);
    unsigned r = u + 0x7FFFu + ((u >> 16) & 1u);
    return (unsigned short)(r >> 16);
}

// ---------- transpose + convert: src (R x C f32) -> dst (C x R bf16) ----------
__global__ __launch_bounds__(256) void transpose_cvt(
    const float* __restrict__ src, unsigned short* __restrict__ dst,
    int R, int C) {
    __shared__ float tile[32][33];
    int c0 = blockIdx.x * 32;
    int r0 = blockIdx.y * 32;
    int tx = threadIdx.x & 31, ty = threadIdx.x >> 5;  // ty 0..7
#pragma unroll
    for (int i = ty; i < 32; i += 8)
        tile[i][tx] = src[(size_t)(r0 + i) * C + c0 + tx];
    __syncthreads();
#pragma unroll
    for (int i = ty; i < 32; i += 8)
        dst[(size_t)(c0 + i) * R + r0 + tx] = f2bf_rne(tile[tx][i]);
}

// ---------- gather embeddings to bf16 (row-major M x D) ----------
__global__ void gather_emb(const float* __restrict__ emb,
                           const long long* __restrict__ ids,
                           unsigned short* __restrict__ X) {
    int row = blockIdx.x;  // 0..2047
    long long t = ids[row];
    const float* src = emb + (size_t)t * DD;
    for (int c = threadIdx.x; c < DD; c += blockDim.x)
        X[(size_t)row * DD + c] = f2bf_rne(src[c]);
}

// ---------- TDM tile load: 128 rows x 32 k bf16, row stride = K elems,
// ---------- LDS rows padded to 40 halfwords (pad 16B every 64B stored) ----------
#if USE_TDM
__device__ __forceinline__ void tdm_load_tile(unsigned lds_byte,
                                              const unsigned short* gsrc,
                                              int row_stride_elems) {
    unsigned long long ga = (unsigned long long)(size_t)gsrc;
    u32x4 g0;
    g0.x = 1u;                                   // count=1, user descriptor
    g0.y = lds_byte;                             // lds_addr (bytes)
    g0.z = (unsigned)ga;                         // global_addr[31:0]
    g0.w = ((unsigned)(ga >> 32) & 0x01FFFFFFu)  // global_addr[56:32]
           | 0x80000000u;                        // type=2 ("image")
    i32x8 g1;
    g1[0] = (int)((1u << 16)      // data_size = 2 bytes
                  | (1u << 20)    // pad_enable
                  | (3u << 22)    // pad_interval: 16 DWORDs (64B)
                  | (3u << 25));  // pad_amount: 4 DWORDs (16B)
    g1[1] = (int)(32u << 16);     // tensor_dim0 = 32
    g1[2] = (int)(128u << 16);    // tensor_dim1 = 128
    g1[3] = (int)(32u << 16);     // tile_dim0 = 32
    g1[4] = 128;                  // tile_dim1 = 128, tile_dim2 = 0
    g1[5] = row_stride_elems;     // tensor_dim0_stride[31:0]
    g1[6] = 0;
    g1[7] = 0;
    i32x4 z4 = {0, 0, 0, 0};
#if __clang_major__ >= 23
    i32x8 z8 = {};
    __builtin_amdgcn_tensor_load_to_lds(g0, g1, z4, z4, z8, 0);
#else
    __builtin_amdgcn_tensor_load_to_lds(g0, g1, z4, z4, 0);
#endif
}
#endif

// ---------- stage one double-buffer slot (A tile + B tile) ----------
__device__ __forceinline__ void stage_tiles(unsigned short* lds,
                                            unsigned lds_byte_base, int buf,
                                            const unsigned short* Atile,
                                            const unsigned short* Btile, int K,
                                            int tid) {
#if USE_TDM
    if (tid < 32) {  // wave 0 drives the TDM
        tdm_load_tile(lds_byte_base + (unsigned)buf * 20480u, Atile, K);
        tdm_load_tile(lds_byte_base + (unsigned)buf * 20480u + 10240u, Btile, K);
    }
#else
    (void)lds_byte_base;
    unsigned short* dA = lds + buf * 10240;
    unsigned short* dB = dA + 5120;
#pragma unroll
    for (int c = tid; c < 512; c += 256) {
        int row = c >> 2, kc = (c & 3) << 3;
        *(uint4*)(dA + row * 40 + kc) =
            *(const uint4*)(Atile + (size_t)row * K + kc);
        *(uint4*)(dB + row * 40 + kc) =
            *(const uint4*)(Btile + (size_t)row * K + kc);
    }
#endif
}

__device__ __forceinline__ void stage_wait(int tid) {
#if USE_TDM
    if (tid < 32) __builtin_amdgcn_s_wait_tensorcnt(0);
#endif
    __syncthreads();
}

// ---------- WMMA GEMM: C = A(MxK,bf16,row) * B^T(NxK,bf16,row) + bias ----------
// mode 0: store f32 to Cf.   mode 1: gelu(tanh approx) then bf16 to Cb.
// Requires M%128==0, N%128==0, K%32==0.
__global__ __launch_bounds__(256) void gemm_wmma_bf16(
    const unsigned short* __restrict__ A, const unsigned short* __restrict__ B,
    const float* __restrict__ bias, float* __restrict__ Cf,
    unsigned short* __restrict__ Cb, int M, int N, int K, int mode) {
    // 2 buffers x (A: 128x40 + B: 128x40) halfwords = 40960 bytes
    __shared__ __align__(16) unsigned short lds[20480];

    const int tid  = threadIdx.x;
    const int m0   = blockIdx.y * 128;
    const int n0   = blockIdx.x * 128;
    const int w    = tid >> 5;
    const int lane = tid & 31;
    const int lr   = lane & 15;
    const int hl   = lane >> 4;
    const int mw   = w & 3;   // 4 waves along M
    const int nw   = w >> 2;  // 2 waves along N

    const unsigned ldsb = (unsigned)(size_t)(&lds[0]);  // LDS byte offset
    const unsigned short* Abase = A + (size_t)m0 * K;
    const unsigned short* Bbase = B + (size_t)n0 * K;

    v8f acc[2][4] = {};

    stage_tiles(lds, ldsb, 0, Abase, Bbase, K, tid);
    stage_wait(tid);

    const int nT = K / 32;
    for (int t = 0; t < nT; ++t) {
        const int cur = t & 1;
        if (t + 1 < nT)  // prefetch next K-tile into the other buffer
            stage_tiles(lds, ldsb, cur ^ 1, Abase + (t + 1) * 32,
                        Bbase + (t + 1) * 32, K, tid);

        const unsigned short* lA = lds + cur * 10240;
        const unsigned short* lB = lA + 5120;

        // A frags: lane<16 -> row lr, K {0..7,16..23}; lane>=16 -> K {8..15,24..31}
        v16bf afrag[2];
#pragma unroll
        for (int i = 0; i < 2; i++) {
            int base = (mw * 32 + i * 16 + lr) * 40 + hl * 8;
            union { v16bf v; uint4 q[2]; } u;
            u.q[0] = *(const uint4*)(lA + base);
            u.q[1] = *(const uint4*)(lA + base + 16);
            afrag[i] = u.v;
        }
        // B frags: lane<16 -> col lr, K {0..7,16..23}; lane>=16 -> K {8..15,24..31}
        v16bf bfrag[4];
#pragma unroll
        for (int j = 0; j < 4; j++) {
            int base = (nw * 64 + j * 16 + lr) * 40 + hl * 8;
            union { v16bf v; uint4 q[2]; } u;
            u.q[0] = *(const uint4*)(lB + base);
            u.q[1] = *(const uint4*)(lB + base + 16);
            bfrag[j] = u.v;
        }
#pragma unroll
        for (int i = 0; i < 2; i++)
#pragma unroll
            for (int j = 0; j < 4; j++)
                acc[i][j] = __builtin_amdgcn_wmma_f32_16x16x32_bf16(
                    false, afrag[i], false, bfrag[j], (short)0, acc[i][j],
                    false, false);

        stage_wait(tid);
    }

    // epilogue: C VGPR r holds M = r (lanes 0-15) / r+8 (lanes 16-31), N = lr
#pragma unroll
    for (int i = 0; i < 2; i++) {
#pragma unroll
        for (int j = 0; j < 4; j++) {
            int gn = n0 + nw * 64 + j * 16 + lr;
            float bv = bias[gn];
#pragma unroll
            for (int r = 0; r < 8; r++) {
                int gm = m0 + mw * 32 + i * 16 + hl * 8 + r;
                float v = acc[i][j][r] + bv;
                if (mode == 1) {
                    float x = v;
                    float t2 = 0.7978845608028654f * (x + 0.044715f * x * x * x);
                    v = 0.5f * x * (1.0f + tanhf(t2));
                    Cb[(size_t)gm * N + gn] = f2bf_rne(v);
                } else {
                    Cf[(size_t)gm * N + gn] = v;
                }
            }
        }
    }
}

// ---------- per-row log-softmax in place (one block per row) ----------
__global__ __launch_bounds__(256) void row_logsoftmax(float* __restrict__ out) {
    float* row = out + (size_t)blockIdx.x * VOCAB;
    __shared__ float sm[256], ss[256];
    __shared__ float s_lse;
    float m = -3.4e38f, s = 0.0f;
    for (int v = threadIdx.x; v < VOCAB; v += 256) {
        float x = row[v];
        if (x > m) { s = s * expf(m - x) + 1.0f; m = x; }
        else       { s += expf(x - m); }
    }
    sm[threadIdx.x] = m; ss[threadIdx.x] = s;
    __syncthreads();
    if (threadIdx.x == 0) {
        float M = -3.4e38f, S = 0.0f;
        for (int t = 0; t < 256; t++) {
            float mt = sm[t], st = ss[t];
            if (mt > M) { S = S * expf(M - mt) + st; M = mt; }
            else        { S += st * expf(mt - M); }
        }
        s_lse = M + logf(S);
    }
    __syncthreads();
    float lse = s_lse;
    for (int v = threadIdx.x; v < VOCAB; v += 256) row[v] -= lse;
}

// ---------- sidecar hash probe (tiny, 4 lanes do real work) ----------
__global__ void sidecar_probe(
    const long long* __restrict__ ids, const int* __restrict__ attn,
    const int* __restrict__ occ, const int* __restrict__ sord,
    const unsigned long long* __restrict__ sh1,
    const unsigned long long* __restrict__ sh2,
    const long long* __restrict__ soff, const long long* __restrict__ slen,
    const long long* __restrict__ stot, const long long* __restrict__ stopc,
    const long long* __restrict__ ptok, const long long* __restrict__ pcnt,
    int paylen, Sidecar* __restrict__ sc) {
    int b = threadIdx.x;
    if (b >= BB) return;
    int vl = 0;
    for (int s = 0; s < SS; s++) vl += attn[b * SS + s];

    const int orders[5] = {8, 7, 6, 5, 4};
    int sel = -1;
    long long sidx_sel = 0;
    for (int oi = 0; oi < 5; oi++) {
        int o = orders[oi];
        unsigned long long h1 = 0ull, h2 = 0ull;
        for (int j = 0; j < o; j++) {
            int pos = vl - o + j;
            if (pos < 0) pos = 0;
            if (pos > SS - 1) pos = SS - 1;
            unsigned long long x = (unsigned long long)(ids[b * SS + pos] + 1);
            h1 = h1 * 1099511628211ull + x;
            h2 = h2 * 6364136223846793005ull + x;
        }
        unsigned long long x = h1 ^ (h2 << 1) ^
            ((unsigned long long)(o + 1) * 11400714819323198485ull);
        x ^= x >> 33; x *= 18397679294719823053ull;
        x ^= x >> 33; x *= 14181476777654086739ull;
        x ^= x >> 33;
        long long idx0 = (long long)(x & (unsigned long long)MASKI);

        int fm = 16, fe = 16;
        for (int p = 0; p < 16; p++) {
            long long idx = (idx0 + p) & MASKI;
            int oc = occ[idx];
            if (oc == 0 && fe == 16) fe = p;
            if (oc > 0 && fm == 16 && sord[idx] == o &&
                sh1[idx] == h1 && sh2[idx] == h2) fm = p;
        }
        long long sidx = (idx0 + (fm < 15 ? fm : 15)) & MASKI;
        long long len = slen[sidx];
        long long bytes = (long long)(fm + 1) * 48 + len * 12;
        int found = (fm < fe) && (fm < 16) && (bytes <= 256) && (o <= vl);
        if (found && sel < 0) { sel = oi; sidx_sel = sidx; }
    }

    Sidecar r = {};
    r.valid_len = vl;
    r.any_hit = (sel >= 0);
    if (r.any_hit) {
        long long tot = stot[sidx_sel], topc = stopc[sidx_sel];
        long long off = soff[sidx_sel], len = slen[sidx_sel];
        r.stot = tot;
        r.conf = (float)topc / fmaxf((float)tot, 1.0f);
        r.slen = (int)len;
        float c[8]; float csum = 0.0f;
        for (int j = 0; j < 8; j++) {
            long long pidx = off + j;
            if (pidx < 0) pidx = 0;
            if (pidx > paylen - 1) pidx = paylen - 1;
            r.ptok[j] = (int)ptok[pidx];
            c[j] = (j < len) ? (float)pcnt[pidx] : 0.0f;
            csum += c[j];
        }
        float inv = 1.0f / fmaxf(csum, 1e-30f);
        for (int j = 0; j < 8; j++) r.contrib[j] = 0.7f * c[j] * inv;  // W_SIDE
        r.top_tok = r.ptok[0];
    }
    sc[b] = r;
}

__device__ __forceinline__ void ins8(float* a, float v) {
    if (v <= a[7]) return;
    a[7] = v;
    for (int i = 7; i > 0 && a[i] > a[i - 1]; --i) {
        float t = a[i]; a[i] = a[i - 1]; a[i - 1] = t;
    }
}

// ---------- blend sidecar distribution into last position ----------
__global__ __launch_bounds__(256) void sidecar_blend(
    float* __restrict__ out, const Sidecar* __restrict__ sc) {
    int b = blockIdx.x;
    Sidecar s = sc[b];
    int last = s.valid_len - 1;
    if (last < 0) last = 0;
    if (last > SS - 1) last = SS - 1;
    float* row = out + ((size_t)b * SS + last) * VOCAB;

    __shared__ float topbuf[256 * 8];
    __shared__ int s_fused;

    float loc[8];
#pragma unroll
    for (int i = 0; i < 8; i++) loc[i] = -1.0f;
    for (int v = threadIdx.x; v < VOCAB; v += 256) ins8(loc, expf(row[v]));
#pragma unroll
    for (int i = 0; i < 8; i++) topbuf[threadIdx.x * 8 + i] = loc[i];
    __syncthreads();

    if (threadIdx.x == 0) {
        float m[8];
        for (int i = 0; i < 8; i++) m[i] = -1.0f;
        for (int i = 0; i < 256 * 8; i++) ins8(m, topbuf[i]);
        float kth = m[7];  // 8th largest prob
        int tt = s.top_tok;
        if (tt < 0) tt = 0;
        if (tt > VOCAB - 1) tt = VOCAB - 1;
        float p_top = expf(row[tt]);
        s_fused = s.any_hit && (s.valid_len >= 4) && (s.stot >= 2) &&
                  (s.conf >= 0.55f) && (p_top >= 0.02f) && (p_top >= kth);
    }
    __syncthreads();

    if (s_fused) {
        for (int v = threadIdx.x; v < VOCAB; v += 256) {
            float p  = expf(row[v]);
            float pn = 0.3f * p;  // (1 - W_SIDE) * p
#pragma unroll
            for (int j = 0; j < 8; j++)
                if (s.ptok[j] == v) pn += s.contrib[j];
            row[v] = logf(fmaxf(pn, 1e-30f));
        }
    }
}

// ---------- launcher ----------
extern "C" void kernel_launch(void* const* d_in, const int* in_sizes, int n_in,
                              void* d_out, int out_size, void* d_ws,
                              size_t ws_size, hipStream_t stream) {
    (void)n_in; (void)out_size; (void)ws_size;
    const long long*          input_ids = (const long long*)d_in[0];
    const int*                attn      = (const int*)d_in[1];
    const float*              emb       = (const float*)d_in[2];
    const float*              W1        = (const float*)d_in[3];
    const float*              b1        = (const float*)d_in[4];
    const float*              W2        = (const float*)d_in[5];
    const float*              b2        = (const float*)d_in[6];
    const int*                slot_occ  = (const int*)d_in[7];
    const int*                slot_ord  = (const int*)d_in[8];
    const unsigned long long* slot_h1   = (const unsigned long long*)d_in[9];
    const unsigned long long* slot_h2   = (const unsigned long long*)d_in[10];
    const long long*          slot_off  = (const long long*)d_in[11];
    const long long*          slot_len  = (const long long*)d_in[12];
    const long long*          slot_tot  = (const long long*)d_in[13];
    const long long*          slot_top  = (const long long*)d_in[14];
    const long long*          pay_tok   = (const long long*)d_in[15];
    const long long*          pay_cnt   = (const long long*)d_in[16];
    int paylen = in_sizes[15];
    float* out = (float*)d_out;

    // workspace layout: transposed bf16 weights, bf16 activations, sidecar
    unsigned short* wsu = (unsigned short*)d_ws;
    unsigned short* W1t = wsu;                               // 1024 x 1024  (NxK)
    unsigned short* W2t = W1t + (size_t)DD * DD;             // 32000 x 1024 (NxK)
    unsigned short* Xb  = W2t + (size_t)DD * VOCAB;          // 2048 x 1024
    unsigned short* Hb  = Xb + (size_t)BB * SS * DD;         // 2048 x 1024
    Sidecar*        sc  = (Sidecar*)(Hb + (size_t)BB * SS * DD);

    const int M = BB * SS;  // 2048

    // 1. transpose+convert weights to N x K bf16
    {
        dim3 g(DD / 32, DD / 32);
        transpose_cvt<<<g, 256, 0, stream>>>(W1, W1t, DD, DD);
    }
    {
        dim3 g(VOCAB / 32, DD / 32);
        transpose_cvt<<<g, 256, 0, stream>>>(W2, W2t, DD, VOCAB);
    }
    // 2. embedding gather -> bf16 X
    gather_emb<<<M, 256, 0, stream>>>(emb, input_ids, Xb);

    // 3. GEMM1 + GELU -> bf16 H   (2048 x 1024 x 1024)
    {
        dim3 g(DD / 128, M / 128);
        gemm_wmma_bf16<<<g, 256, 0, stream>>>(Xb, W1t, b1, nullptr, Hb,
                                              M, DD, DD, 1);
    }
    // 4. GEMM2 -> f32 logits in d_out   (2048 x 32000 x 1024)
    {
        dim3 g(VOCAB / 128, M / 128);
        gemm_wmma_bf16<<<g, 256, 0, stream>>>(Hb, W2t, b2, out, nullptr,
                                              M, VOCAB, DD, 0);
    }
    // 5. log-softmax in place, one block per row
    row_logsoftmax<<<M, 256, 0, stream>>>(out);

    // 6. sidecar probe (tiny)
    sidecar_probe<<<1, 32, 0, stream>>>(input_ids, attn, slot_occ, slot_ord,
                                        slot_h1, slot_h2, slot_off, slot_len,
                                        slot_tot, slot_top, pay_tok, pay_cnt,
                                        paylen, sc);
    // 7. blend into last positions
    sidecar_blend<<<BB, 256, 0, stream>>>(out, sc);
}
